// ChannelMixing_37821482008735
// MI455X (gfx1250) — matrix operand order repaired and checked by reading, not compiled, not run on hardware
//
#include <hip/hip_runtime.h>
#include <stdint.h>
#include <stddef.h>

#define LEN   4096
#define DIM   2048
#define CHUNK 256
#define NCH   (LEN / CHUNK)   // 16

typedef __bf16 v16bf __attribute__((ext_vector_type(16)));
typedef float  v8f   __attribute__((ext_vector_type(8)));

// ---------------------------------------------------------------- helpers
__device__ __forceinline__ unsigned short f2bf(float f) {
    union { float f; uint32_t u; } v; v.f = f;
    uint32_t u = v.u;
    return (unsigned short)((u + 0x7FFFu + ((u >> 16) & 1u)) >> 16); // RNE
}
__device__ __forceinline__ float sigm(float v) {
    return 1.0f / (1.0f + __expf(-v));
}

// CDNA5 async global->LDS copy (ASYNCcnt-tracked DMA, no VGPR round trip).
// vdst = 32-bit LDS byte address, vaddr = 64-bit global address.
__device__ __forceinline__ void async_ld128(const void* lds, const void* gaddr) {
    asm volatile("global_load_async_to_lds_b128 %0, %1, off"
                 :: "v"((unsigned)(size_t)lds),
                    "v"((unsigned long long)(size_t)gaddr)
                 : "memory");
}
__device__ __forceinline__ void wait_async0() {
    asm volatile("s_wait_asynccnt 0x0" ::: "memory");
}

// ---------------------------------------------------------------- fp32 -> bf16 convert
__global__ void cvt_bf16(const float* __restrict__ src,
                         unsigned short* __restrict__ dst, int n) {
    int i = (blockIdx.x * blockDim.x + threadIdx.x) * 4;
    if (i + 3 < n) {
        float4 f = *(const float4*)(src + i);
        ushort4 o;
        o.x = f2bf(f.x); o.y = f2bf(f.y); o.z = f2bf(f.z); o.w = f2bf(f.w);
        *(ushort4*)(dst + i) = o;
    }
}

// ---------------------------------------------------------------- blocked scan, pass 1
__global__ void scan_partial(const float* __restrict__ x,
                             const float* __restrict__ mix_k,
                             const float* __restrict__ mix_r,
                             float* __restrict__ Ek, float* __restrict__ Er) {
    int g = blockIdx.x * blockDim.x + threadIdx.x;   // 0 .. DIM*NCH-1
    int d = g & (DIM - 1);
    int c = g >> 11;                                 // g / DIM
    float mk = sigm(mix_k[d]), mr = sigm(mix_r[d]);
    float ak = 1.0f - mk, ar = 1.0f - mr;
    float lk = 0.0f, lr = 0.0f;
    const float* xp = x + (size_t)c * CHUNK * DIM + d;
    #pragma unroll 4
    for (int t = 0; t < CHUNK; ++t) {
        float xv = xp[(size_t)t * DIM];
        lk = fmaf(mk, lk, ak * xv);
        lr = fmaf(mr, lr, ar * xv);
    }
    Ek[c * DIM + d] = lk;
    Er[c * DIM + d] = lr;
}

// pass 2: sequential combine over the 16 chunks -> state entering each chunk.
__global__ void scan_combine(const float* __restrict__ mix_k,
                             const float* __restrict__ mix_r,
                             const float* __restrict__ lastk,
                             const float* __restrict__ lastr,
                             const float* __restrict__ Ek, const float* __restrict__ Er,
                             float* __restrict__ Hk, float* __restrict__ Hr) {
    int d = blockIdx.x * blockDim.x + threadIdx.x;   // 0..DIM-1
    float mk = sigm(mix_k[d]), mr = sigm(mix_r[d]);
    float mkC = mk, mrC = mr;                        // m^CHUNK via 8 squarings (CHUNK=256)
    #pragma unroll
    for (int i = 0; i < 8; ++i) { mkC *= mkC; mrC *= mrC; }
    float hk = lastk[d], hr = lastr[d];
    #pragma unroll
    for (int c = 0; c < NCH; ++c) {
        Hk[c * DIM + d] = hk;
        Hr[c * DIM + d] = hr;
        hk = fmaf(mkC, hk, Ek[c * DIM + d]);
        hr = fmaf(mrC, hr, Er[c * DIM + d]);
    }
}

// pass 3: recompute local scan with correction m^(t-t0+1)*h_start, emit bf16 X tiles.
__global__ void scan_final(const float* __restrict__ x,
                           const float* __restrict__ mix_k,
                           const float* __restrict__ mix_r,
                           const float* __restrict__ Hk, const float* __restrict__ Hr,
                           unsigned short* __restrict__ Xk,
                           unsigned short* __restrict__ Xr) {
    int g = blockIdx.x * blockDim.x + threadIdx.x;
    int d = g & (DIM - 1);
    int c = g >> 11;
    float mk = sigm(mix_k[d]), mr = sigm(mix_r[d]);
    float ak = 1.0f - mk, ar = 1.0f - mr;
    float hk0 = Hk[c * DIM + d], hr0 = Hr[c * DIM + d];
    float fk = mk, fr = mr, lk = 0.0f, lr = 0.0f;
    size_t base = (size_t)c * CHUNK * DIM + d;
    #pragma unroll 4
    for (int t = 0; t < CHUNK; ++t) {
        float xv = x[base + (size_t)t * DIM];
        lk = fmaf(mk, lk, ak * xv);
        lr = fmaf(mr, lr, ar * xv);
        Xk[base + (size_t)t * DIM] = f2bf(fmaf(fk, hk0, lk));
        Xr[base + (size_t)t * DIM] = f2bf(fmaf(fr, hr0, lr));
        fk *= mk; fr *= mr;
    }
}

// ---------------------------------------------------------------- bf16 NT GEMM + fused epilogue
// C[M,N] = A[M,K](row major) * B[N,K](row major)^T  via v_wmma_f32_16x16x32_bf16
// Block tile 128x256, BK=32, 8 waves (2x4), wave tile 64x64 (16 wmma / 16 ds_load per step).
// Global->LDS staging uses async-to-LDS DMA, double-buffered.
// MODE 0: out_bf = bf16(relu(acc)^2)   MODE 1: out_f = sigmoid(acc)   MODE 2: out_f = acc*aux
#define BM 128
#define BN 256
#define BK 32
#define LDSS 40     // padded row stride (elements); 80B keeps 16B alignment

template <int MODE>
__global__ __launch_bounds__(256)
void gemm_nt(const unsigned short* __restrict__ A,
             const unsigned short* __restrict__ B,
             unsigned short* __restrict__ outBf,
             float* __restrict__ outF,
             const float* __restrict__ aux,
             int Kdim, int Ndim) {
    __shared__ __align__(16) unsigned short sA[2][BM * LDSS];
    __shared__ __align__(16) unsigned short sB[2][BN * LDSS];

    const int tid  = threadIdx.x;
    const int wave = tid >> 5;
    const int lane = tid & 31;
    const int half = lane >> 4;     // selects K sub-chunks per ISA layout
    const int r    = lane & 15;

    const int waveM = wave >> 2;    // 2 waves in M
    const int waveN = wave & 3;     // 4 waves in N
    const int blockM = blockIdx.y * BM;
    const int blockN = blockIdx.x * BN;

    const int lr = tid >> 2;        // cooperative-load row 0..63
    const int lc = (tid & 3) * 8;   // 16B chunk within 32-element K slab

    v8f acc[4][4];
    #pragma unroll
    for (int mi = 0; mi < 4; ++mi)
        #pragma unroll
        for (int ni = 0; ni < 4; ++ni)
            acc[mi][ni] = (v8f)0.0f;

    const unsigned short* gA = A + (size_t)(blockM + lr) * Kdim + lc;
    const unsigned short* gB = B + (size_t)(blockN + lr) * Kdim + lc;

    auto issue_tile = [&](int buf, int k0) {
        // A: 128 rows -> 2 chunks/thread; B: 256 rows -> 4 chunks/thread
        async_ld128(&sA[buf][lr * LDSS + lc],        gA + k0);
        async_ld128(&sA[buf][(lr + 64) * LDSS + lc], gA + k0 + (size_t)64 * Kdim);
        #pragma unroll
        for (int i = 0; i < 4; ++i)
            async_ld128(&sB[buf][(lr + 64 * i) * LDSS + lc],
                        gB + k0 + (size_t)(64 * i) * Kdim);
    };

    issue_tile(0, 0);
    wait_async0();
    __syncthreads();

    int cur = 0;
    for (int k0 = 0; k0 < Kdim; k0 += BK) {
        if (k0 + BK < Kdim) issue_tile(cur ^ 1, k0 + BK);

        union Frag { uint4 u[2]; v16bf h; };
        Frag a[4], b[4];
        // A frag (16x32): lane<16 -> K{0..7,16..23}, lane>=16 -> K{8..15,24..31}
        #pragma unroll
        for (int mi = 0; mi < 4; ++mi) {
            const unsigned short* p = &sA[cur][(waveM * 64 + mi * 16 + r) * LDSS + half * 8];
            a[mi].u[0] = *(const uint4*)p;
            a[mi].u[1] = *(const uint4*)(p + 16);
        }
        // B frag (32x16): lane<16 -> K{0..7,16..23}, lane>=16 -> K{8..15,24..31} of column r
        #pragma unroll
        for (int ni = 0; ni < 4; ++ni) {
            const unsigned short* p = &sB[cur][(waveN * 64 + ni * 16 + r) * LDSS + half * 8];
            b[ni].u[0] = *(const uint4*)p;
            b[ni].u[1] = *(const uint4*)(p + 16);
        }
        #pragma unroll
        for (int mi = 0; mi < 4; ++mi)
            #pragma unroll
            for (int ni = 0; ni < 4; ++ni)
                acc[mi][ni] = __builtin_amdgcn_wmma_f32_16x16x32_bf16(
                    false, a[mi].h, false, b[ni].h, (short)0, acc[mi][ni], false, false);

        wait_async0();     // next buffer's DMA complete (this wave's issues)
        __syncthreads();   // visible to all waves; cur fully consumed by all
        cur ^= 1;
    }

    // epilogue: C/D layout -> VGPR j holds M = j + 8*half, N = lane%16
    #pragma unroll
    for (int mi = 0; mi < 4; ++mi) {
        #pragma unroll
        for (int ni = 0; ni < 4; ++ni) {
            int gCol = blockN + waveN * 64 + ni * 16 + r;
            #pragma unroll
            for (int j = 0; j < 8; ++j) {
                int gRow   = blockM + waveM * 64 + mi * 16 + half * 8 + j;
                size_t idx = (size_t)gRow * Ndim + gCol;
                float v = acc[mi][ni][j];
                if (MODE == 0) {
                    float t = v > 0.0f ? v : 0.0f;
                    outBf[idx] = f2bf(t * t);
                } else if (MODE == 1) {
                    outF[idx] = sigm(v);
                } else {
                    outF[idx] = v * aux[idx];
                }
            }
        }
    }
}

// ---------------------------------------------------------------- launch
extern "C" void kernel_launch(void* const* d_in, const int* in_sizes, int n_in,
                              void* d_out, int out_size, void* d_ws, size_t ws_size,
                              hipStream_t stream) {
    const float* x     = (const float*)d_in[0];
    const float* Wk    = (const float*)d_in[1];
    const float* Wr    = (const float*)d_in[2];
    const float* Wv    = (const float*)d_in[3];
    const float* mixk  = (const float*)d_in[4];
    const float* mixr  = (const float*)d_in[5];
    const float* lastk = (const float*)d_in[6];
    const float* lastr = (const float*)d_in[7];
    float* out = (float*)d_out;

    const size_t WN = (size_t)DIM * DIM;   // 4M weight elems
    const size_t XN = (size_t)LEN * DIM;   // 8M activation elems
    char* ws = (char*)d_ws;
    unsigned short* WkB = (unsigned short*)(ws);
    unsigned short* WrB = (unsigned short*)(ws + WN * 2);
    unsigned short* WvB = (unsigned short*)(ws + WN * 4);
    unsigned short* Xk  = (unsigned short*)(ws + WN * 6);
    unsigned short* Xr  = (unsigned short*)(ws + WN * 6 + XN * 2);
    unsigned short* Act = (unsigned short*)(ws + WN * 6 + XN * 4);
    float*          S   = (float*)        (ws + WN * 6 + XN * 6);
    float*          Ek  = (float*)        (ws + WN * 6 + XN * 6 + XN * 4);
    float*          Er  = Ek + DIM * NCH;
    float*          Hk  = Er + DIM * NCH;
    float*          Hr  = Hk + DIM * NCH;

    // weights fp32 -> bf16
    cvt_bf16<<<(int)(WN / 1024), 256, 0, stream>>>(Wk, WkB, (int)WN);
    cvt_bf16<<<(int)(WN / 1024), 256, 0, stream>>>(Wr, WrB, (int)WN);
    cvt_bf16<<<(int)(WN / 1024), 256, 0, stream>>>(Wv, WvB, (int)WN);

    // decay-mix scan (== FFT conv in the reference), blocked over time
    scan_partial<<<(DIM * NCH) / 256, 256, 0, stream>>>(x, mixk, mixr, Ek, Er);
    scan_combine<<<DIM / 256, 256, 0, stream>>>(mixk, mixr, lastk, lastr, Ek, Er, Hk, Hr);
    scan_final  <<<(DIM * NCH) / 256, 256, 0, stream>>>(x, mixk, mixr, Hk, Hr, Xk, Xr);

    // three NT GEMMs with fused epilogues
    dim3 grid(DIM / BN, LEN / BM);   // 8 x 32
    gemm_nt<0><<<grid, 256, 0, stream>>>(Xk,  WkB, Act, nullptr, nullptr, DIM, DIM);
    gemm_nt<1><<<grid, 256, 0, stream>>>(Xr,  WrB, nullptr, S,   nullptr, DIM, DIM);
    gemm_nt<2><<<grid, 256, 0, stream>>>(Act, WvB, nullptr, out, S,       DIM, DIM);
}
